// _IPEXAttention_73400991088963
// MI455X (gfx1250) — matrix operand order repaired and checked by reading, not compiled, not run on hardware
//
#include <hip/hip_runtime.h>
#include <hip/hip_bf16.h>

typedef __attribute__((ext_vector_type(16))) _Float16 v16h;
typedef __attribute__((ext_vector_type(8)))  float    v8f;
typedef __attribute__((ext_vector_type(4)))  float    f4v;

#define NUM_HEADS      32
#define KV_HEADS       8
#define GROUPS         4      // NUM_HEADS / KV_HEADS
#define HEAD_DIM       128
#define BLOCK_SIZE     16
#define BLOCKS_PER_SEQ 128
#define NW             4      // waves per workgroup
#define WAVE           32

__global__ __launch_bounds__(NW * WAVE)
void paged_attn_decode_wmma(const float* __restrict__ q,
                            const float* __restrict__ kc,
                            const float* __restrict__ vc,
                            const int*   __restrict__ bt,
                            const int*   __restrict__ lens,
                            float*       __restrict__ out)
{
    const int b    = blockIdx.x / KV_HEADS;
    const int kvh  = blockIdx.x % KV_HEADS;
    const int tid  = threadIdx.x;
    const int w    = tid / WAVE;
    const int lane = tid % WAVE;
    const int nloc = lane & 15;   // column / row-within-half index
    const int hi   = lane >> 4;   // 0: lanes 0-15, 1: lanes 16-31

    __shared__ float    sV[NW][32 * HEAD_DIM];    // async-DMA'd V tile (16 KB/wave)
    __shared__ _Float16 sP[NW][16 * 32];          // P staging (A-fragment source)
    __shared__ float    sO[NW][GROUPS][HEAD_DIM]; // per-wave unnormalized output
    __shared__ float    sM[NW][GROUPS];
    __shared__ float    sL[NW][GROUPS];

    // zero the P staging tile; rows 4..15 stay zero forever (padded WMMA rows)
    for (int i = lane; i < 16 * 32; i += WAVE) sP[w][i] = (_Float16)0.0f;

    const int   len   = lens[b];
    const int   T     = (len + 31) >> 5;          // 32-token context tiles
    const float scale = 0.08838834764831845f;     // 1/sqrt(128)

    // per-lane LDS byte address for the async V copy (lane-contiguous 16B)
    const uint32_t vlds = (uint32_t)(uintptr_t)(&sV[w][0]) + (uint32_t)lane * 16u;

    // ---- Q A-fragments: 16x32 f16 per 32-wide head-dim chunk, scale folded in
    v16h aq[4];
    {
        const int m = nloc;  // A-matrix row
        const float* qp = q + ((size_t)b * NUM_HEADS + (kvh * GROUPS + (m & 3))) * HEAD_DIM;
        #pragma unroll
        for (int c = 0; c < 4; ++c) {
            v16h a;
            #pragma unroll
            for (int e = 0; e < 16; ++e) {
                const int kk = c * 32 + hi * 16 + e;
                const float v = (m < GROUPS) ? qp[kk] * scale : 0.0f;
                a[e] = (_Float16)v;
            }
            aq[c] = a;
        }
    }

    float m_[GROUPS], l_[GROUPS];
    #pragma unroll
    for (int g = 0; g < GROUPS; ++g) { m_[g] = -__builtin_inff(); l_[g] = 0.0f; }
    v8f acc[8] = {};   // O accumulator: 8 chunks of 16 head-dim columns

    for (int t = w; t < T; t += NW) {
        const int blk0 = bt[b * BLOCKS_PER_SEQ + 2 * t];
        const int blk1 = bt[b * BLOCKS_PER_SEQ + 2 * t + 1];
        const size_t o0 = ((size_t)blk0 * KV_HEADS + kvh) * (BLOCK_SIZE * HEAD_DIM);
        const size_t o1 = ((size_t)blk1 * KV_HEADS + kvh) * (BLOCK_SIZE * HEAD_DIM);
        const float* k0 = kc + o0;
        const float* k1 = kc + o1;

        // ---- kick off async DMA of the 32x128 f32 V tile into LDS.
        // Each instruction moves 512B (32 lanes x 16B); INST_OFFSET advances the
        // global and LDS addresses together (ISA 8.x async pseudocode).
        {
            const uint64_t g0 = (uint64_t)(uintptr_t)(vc + o0) + (uint64_t)lane * 16u;
            const uint64_t g1 = (uint64_t)(uintptr_t)(vc + o1) + (uint64_t)lane * 16u;
            const uint32_t l1 = vlds + 16u * 512u;  // tokens 16..31 half of the tile
            #pragma unroll
            for (int i = 0; i < 16; ++i)
                asm volatile("global_load_async_to_lds_b128 %0, %1, off offset:%2"
                             :: "v"(vlds), "v"(g0), "i"(i * 512) : "memory");
            #pragma unroll
            for (int i = 0; i < 16; ++i)
                asm volatile("global_load_async_to_lds_b128 %0, %1, off offset:%2"
                             :: "v"(l1), "v"(g1), "i"(i * 512) : "memory");
        }

        // ---- scores S[16x32] = Q(16x128) . K^T : two 16x16 C tiles
        v8f c0 = {}, c1 = {};
        #pragma unroll
        for (int c = 0; c < 4; ++c) {
            // B 32x16: lanes 0-15 hold contraction rows kk=0..15, lanes 16-31 kk=16..31;
            // each lane reads 16 consecutive head-dim floats of one token (4x b128).
            const f4v* kq0 = (const f4v*)(k0 + nloc * HEAD_DIM + c * 32 + hi * 16);
            const f4v* kq1 = (const f4v*)(k1 + nloc * HEAD_DIM + c * 32 + hi * 16);
            v16h b0, b1;
            #pragma unroll
            for (int qd = 0; qd < 4; ++qd) {
                const f4v x0 = kq0[qd], x1 = kq1[qd];
                #pragma unroll
                for (int u = 0; u < 4; ++u) {
                    b0[qd * 4 + u] = (_Float16)x0[u];
                    b1[qd * 4 + u] = (_Float16)x1[u];
                }
            }
            c0 = __builtin_amdgcn_wmma_f32_16x16x32_f16(false, aq[c], false, b0, (short)0, c0, false, false);
            c1 = __builtin_amdgcn_wmma_f32_16x16x32_f16(false, aq[c], false, b1, (short)0, c1, false, false);
        }

        // ---- online softmax over the GROUPS real rows (rows 0..3, lanes 0..15)
        const int pos0 = t * 32 + nloc;
        const int pos1 = pos0 + 16;
        float fac[GROUPS];
        #pragma unroll
        for (int g = 0; g < GROUPS; ++g) {
            float s0 = (pos0 < len) ? c0[g] : -__builtin_inff();
            float s1 = (pos1 < len) ? c1[g] : -__builtin_inff();
            float rm = fmaxf(s0, s1);
            #pragma unroll
            for (int d = 8; d >= 1; d >>= 1) rm = fmaxf(rm, __shfl_xor(rm, d, WAVE));
            const float mn = fmaxf(m_[g], rm);
            fac[g] = __expf(m_[g] - mn);
            const float p0 = __expf(s0 - mn);
            const float p1 = __expf(s1 - mn);
            float rs = p0 + p1;
            #pragma unroll
            for (int d = 8; d >= 1; d >>= 1) rs += __shfl_xor(rs, d, WAVE);
            l_[g] = l_[g] * fac[g] + rs;
            m_[g] = mn;
            if (hi == 0) {
                sP[w][g * 32 + nloc]      = (_Float16)p0;
                sP[w][g * 32 + 16 + nloc] = (_Float16)p1;
            }
        }

        // rescale running accumulator (real rows live at lanes 0-15, r=0..3)
        #pragma unroll
        for (int j = 0; j < 8; ++j)
            #pragma unroll
            for (int r = 0; r < 4; ++r) {
                const float f = (hi == 0) ? fac[r] : 1.0f;
                acc[j][r] *= f;
            }

        // ---- re-fragment P from LDS into A layout (same-wave DS ops are in order)
        __builtin_amdgcn_wave_barrier();
        v16h ap;
        #pragma unroll
        for (int e = 0; e < 16; ++e) {
            const int kctx = hi * 16 + e;
            ap[e] = sP[w][nloc * 32 + kctx];
        }

        // ---- wait for the V DMA, then O += P(16x32) . V(32x16 per chunk)
        asm volatile("s_wait_asynccnt 0" ::: "memory");
        #pragma unroll
        for (int j = 0; j < 8; ++j) {
            v16h bv;
            #pragma unroll
            for (int e = 0; e < 16; ++e)
                bv[e] = (_Float16)sV[w][(hi * 16 + e) * HEAD_DIM + j * 16 + nloc];
            acc[j] = __builtin_amdgcn_wmma_f32_16x16x32_f16(false, ap, false, bv, (short)0, acc[j], false, false);
        }
    }

    // ---- publish per-wave partials
    if (hi == 0) {
        #pragma unroll
        for (int g = 0; g < GROUPS; ++g)
            #pragma unroll
            for (int j = 0; j < 8; ++j)
                sO[w][g][j * 16 + nloc] = acc[j][g];
        if (nloc == 0) {
            #pragma unroll
            for (int g = 0; g < GROUPS; ++g) { sM[w][g] = m_[g]; sL[w][g] = l_[g]; }
        }
    }
    __syncthreads();

    // ---- flash combine across the NW waves, write output
    for (int idx = tid; idx < GROUPS * HEAD_DIM; idx += NW * WAVE) {
        const int g = idx / HEAD_DIM;
        const int d = idx % HEAD_DIM;
        float mg = -__builtin_inff();
        #pragma unroll
        for (int w2 = 0; w2 < NW; ++w2) mg = fmaxf(mg, sM[w2][g]);
        float lg = 0.0f, val = 0.0f;
        #pragma unroll
        for (int w2 = 0; w2 < NW; ++w2) {
            const float e = __expf(sM[w2][g] - mg);
            lg  += sL[w2][g] * e;
            val += sO[w2][g][d] * e;
        }
        out[(size_t)b * (NUM_HEADS * HEAD_DIM) + (kvh * GROUPS + g) * HEAD_DIM + d] = val / lg;
    }
}

extern "C" void kernel_launch(void* const* d_in, const int* in_sizes, int n_in,
                              void* d_out, int out_size, void* d_ws, size_t ws_size,
                              hipStream_t stream) {
    const float* q    = (const float*)d_in[0];
    const float* kc   = (const float*)d_in[1];
    const float* vc   = (const float*)d_in[2];
    const int*   bt   = (const int*)d_in[3];
    const int*   lens = (const int*)d_in[4];
    float*       out  = (float*)d_out;

    const int bsz = in_sizes[4];           // input_lens has one entry per sequence
    dim3 grid(bsz * KV_HEADS);             // one workgroup per (batch, kv_head)
    dim3 block(NW * WAVE);                 // 4 wave32 waves
    paged_attn_decode_wmma<<<grid, block, 0, stream>>>(q, kc, vc, bt, lens, out);
}
